// ToxicCommentModel_46110768890260
// MI455X (gfx1250) — compile-verified, matrix operand order checked by reading, not run on hardware
//
#include <hip/hip_runtime.h>
#include <math.h>

#define B_  128
#define S_  128
#define E_  300
#define HP_ 150
#define HH_ 30
#define ET_ 200
#define VT_ 50000
#define T_  20

typedef __attribute__((ext_vector_type(2)))  float    v2f;
typedef __attribute__((ext_vector_type(8)))  float    v8f;
typedef __attribute__((ext_vector_type(16))) _Float16 v16h;

// ---------------------------------------------------------------- encoder
// per block: one batch row. max over S tokens, adjacent-pair max, fc1+tanh.
__global__ void enc_kernel(const int* __restrict__ X,
                           const float* __restrict__ src_embed,
                           const float* __restrict__ fc1_w,
                           const float* __restrict__ fc1_b,
                           float* __restrict__ h) {
  __shared__ int   toks[S_];
  __shared__ float mx[E_];
  __shared__ float pooled[HP_];
  const int b = blockIdx.x, tid = threadIdx.x;
  for (int s = tid; s < S_; s += blockDim.x) toks[s] = X[b * S_ + s];
  __syncthreads();
  for (int d = tid; d < E_; d += blockDim.x) {
    float m = -INFINITY;
    for (int s = 0; s < S_; ++s) {
      float v = src_embed[(size_t)toks[s] * E_ + d];
      m = fmaxf(m, v);
    }
    mx[d] = m;
  }
  __syncthreads();
  if (tid < HP_) pooled[tid] = fmaxf(mx[2 * tid], mx[2 * tid + 1]);
  __syncthreads();
  if (tid < HH_) {
    float acc = fc1_b[tid];
    const float* w = fc1_w + tid * HP_;
    for (int j = 0; j < HP_; ++j) acc += pooled[j] * w[j];
    h[b * HH_ + tid] = tanhf(acc);
  }
}

// ---------------------------------------------------------------- x0 / amax init
__global__ void initx_kernel(const int* __restrict__ soi_p,
                             const float* __restrict__ tgt_embed,
                             float* __restrict__ x,
                             unsigned long long* __restrict__ amax) {
  const int b = blockIdx.x, tid = threadIdx.x;
  const int soi = soi_p[0];
  for (int j = tid; j < ET_; j += blockDim.x)
    x[b * ET_ + j] = tgt_embed[(size_t)soi * ET_ + j];
  if (tid == 0) amax[b] = 0ull;
}

// ---------------------------------------------------------------- GRU cell
// one block per batch row; also re-arms the argmax accumulator for this step.
__global__ void gru_kernel(const float* __restrict__ x,
                           const float* __restrict__ w_ih,
                           const float* __restrict__ w_hh,
                           const float* __restrict__ b_ih,
                           const float* __restrict__ b_hh,
                           float* __restrict__ h,
                           unsigned long long* __restrict__ amax) {
  __shared__ float xr[ET_];
  __shared__ float hs[HH_];
  __shared__ float gi[3 * HH_];
  __shared__ float gh[3 * HH_];
  const int b = blockIdx.x, tid = threadIdx.x;
  if (tid == 0) amax[b] = 0ull;                       // reset before fc2 of this step
  for (int j = tid; j < ET_; j += blockDim.x) xr[j] = fmaxf(x[b * ET_ + j], 0.0f); // relu
  if (tid < HH_) hs[tid] = h[b * HH_ + tid];
  __syncthreads();
  if (tid < 3 * HH_) {
    float ai = b_ih[tid];
    const float* wi = w_ih + tid * ET_;
    for (int k = 0; k < ET_; ++k) ai += xr[k] * wi[k];
    gi[tid] = ai;
    float ah = b_hh[tid];
    const float* wh = w_hh + tid * HH_;
    for (int k = 0; k < HH_; ++k) ah += hs[k] * wh[k];
    gh[tid] = ah;
  }
  __syncthreads();
  if (tid < HH_) {
    float r = 1.0f / (1.0f + __expf(-(gi[tid] + gh[tid])));
    float z = 1.0f / (1.0f + __expf(-(gi[HH_ + tid] + gh[HH_ + tid])));
    float n = tanhf(gi[2 * HH_ + tid] + r * gh[2 * HH_ + tid]);
    h[b * HH_ + tid] = (1.0f - z) * n + z * hs[tid];
  }
}

// ---------------------------------------------------------------- fc2 (WMMA) + tanh + fused argmax
// grid: 3125 blocks (one 16-col N tile each), 8 waves/block (one 16-row M tile each).
__global__ __launch_bounds__(256) void fc2_kernel(
    const float* __restrict__ h,
    const float* __restrict__ fc2_w,     // [VT_, HH_] row-major => B[k][n] = fc2_w[n*HH_+k]
    const float* __restrict__ fc2_b,
    float* __restrict__ out,             // [B_, T_, VT_]
    unsigned long long* __restrict__ amax,
    int t) {
  const int lane = threadIdx.x & 31;
  const int wave = threadIdx.x >> 5;
  const int m0   = wave << 4;            // M tile base (0..112)
  const int n0   = blockIdx.x << 4;      // N tile base
  const int lh   = lane & 15;
  const int half = lane >> 4;            // 0: lanes 0-15, 1: lanes 16-31
  const int n    = n0 + lh;

  const float* __restrict__ hrow = h + (m0 + lh) * HH_;
  const float* __restrict__ wrow = fc2_w + (size_t)n * HH_;

  v8f c = {0.f, 0.f, 0.f, 0.f, 0.f, 0.f, 0.f, 0.f};

#if __has_builtin(__builtin_amdgcn_wmma_f32_16x16x4_f32)
  // exact fp32 path: 8 chained 16x16x4 WMMAs cover K=0..31 (K=30..31 zero-padded)
  #pragma unroll
  for (int k0 = 0; k0 < 32; k0 += 4) {
    const int ka = k0 + half * 2;        // A lanes 0-15 hold K=k0,k0+1; lanes 16-31 K=k0+2,k0+3
    const int kb = ka + 1;
    v2f a, bf;
    float ax = hrow[ka < HH_ ? ka : 0];  a[0] = (ka < HH_) ? ax : 0.0f;
    float ay = hrow[kb < HH_ ? kb : 0];  a[1] = (kb < HH_) ? ay : 0.0f;
    float bx = wrow[ka < HH_ ? ka : 0];  bf[0] = (ka < HH_) ? bx : 0.0f;
    float by = wrow[kb < HH_ ? kb : 0];  bf[1] = (kb < HH_) ? by : 0.0f;
    c = __builtin_amdgcn_wmma_f32_16x16x4_f32(false, a, false, bf, (short)0, c,
                                              false, false);
  }
#else
  // fallback: one f16 16x16x32 WMMA (K=30 zero-padded to 32), fp32 accumulate
  v16h a, bf;
  #pragma unroll
  for (int e = 0; e < 16; ++e) {
    const int v = e >> 1, s = e & 1;
    const int k = ((v >> 2) << 4) + ((v & 3) << 1) + s + (half << 3);
    a[e]  = (k < HH_) ? (_Float16)hrow[k < HH_ ? k : 0] : (_Float16)0.0f;
    bf[e] = (k < HH_) ? (_Float16)wrow[k < HH_ ? k : 0] : (_Float16)0.0f;
  }
  c = __builtin_amdgcn_wmma_f32_16x16x32_f16(false, a, false, bf, (short)0, c,
                                             false, false);
#endif

  const float bias = fc2_b[n];
  #pragma unroll
  for (int r = 0; r < 8; ++r) {
    const int m = m0 + (half << 3) + r;                // C/D: vgpr r -> row m0+r (+8 upper half)
    const float val = tanhf(c[r] + bias);
    out[(size_t)m * ((size_t)T_ * VT_) + (size_t)t * VT_ + n] = val;

    // sortable pack: higher val wins; ties -> smaller n wins (jnp.argmax semantics)
    unsigned u = __float_as_uint(val);
    u = (u & 0x80000000u) ? ~u : (u | 0x80000000u);
    unsigned long long key =
        ((unsigned long long)u << 32) | (unsigned long long)(0xFFFFFFFFu - (unsigned)n);
    #pragma unroll
    for (int msk = 1; msk <= 8; msk <<= 1) {
      unsigned long long o = __shfl_xor(key, msk, 32);
      if (o > key) key = o;
    }
    if (lh == 0) atomicMax(&amax[m], key);
  }
}

// ---------------------------------------------------------------- greedy re-embed
__global__ void gather_kernel(const float* __restrict__ tgt_embed,
                              float* __restrict__ x,
                              const unsigned long long* __restrict__ amax) {
  const int b = blockIdx.x, tid = threadIdx.x;
  const unsigned long long key = amax[b];
  const int idx = (int)(0xFFFFFFFFu - (unsigned)(key & 0xFFFFFFFFull));
  for (int j = tid; j < ET_; j += blockDim.x)
    x[b * ET_ + j] = tgt_embed[(size_t)idx * ET_ + j];
}

// ---------------------------------------------------------------- launcher
extern "C" void kernel_launch(void* const* d_in, const int* in_sizes, int n_in,
                              void* d_out, int out_size, void* d_ws, size_t ws_size,
                              hipStream_t stream) {
  (void)in_sizes; (void)n_in; (void)out_size; (void)ws_size;
  const int*   X         = (const int*)d_in[0];
  const int*   soi       = (const int*)d_in[1];
  const float* src_embed = (const float*)d_in[2];
  const float* tgt_embed = (const float*)d_in[3];
  const float* fc1_w     = (const float*)d_in[4];
  const float* fc1_b     = (const float*)d_in[5];
  const float* w_ih      = (const float*)d_in[6];
  const float* w_hh      = (const float*)d_in[7];
  const float* b_ih      = (const float*)d_in[8];
  const float* b_hh      = (const float*)d_in[9];
  const float* fc2_w     = (const float*)d_in[10];
  const float* fc2_b     = (const float*)d_in[11];
  float* out = (float*)d_out;

  char* ws = (char*)d_ws;
  float* h = (float*)ws;                                   // 128*30 f32
  float* x = (float*)(ws + 16384);                         // 128*200 f32
  unsigned long long* amax = (unsigned long long*)(ws + 16384 + 102400); // 128 u64

  enc_kernel  <<<B_, 256, 0, stream>>>(X, src_embed, fc1_w, fc1_b, h);
  initx_kernel<<<B_, 256, 0, stream>>>(soi, tgt_embed, x, amax);

  for (int t = 0; t < T_; ++t) {
    gru_kernel<<<B_, 128, 0, stream>>>(x, w_ih, w_hh, b_ih, b_hh, h, amax);
    fc2_kernel<<<VT_ / 16, 256, 0, stream>>>(h, fc2_w, fc2_b, out, amax, t);
    if (t + 1 < T_)
      gather_kernel<<<B_, 256, 0, stream>>>(tgt_embed, x, amax);
  }
}
